// AMTreeGRU_59141699666273
// MI455X (gfx1250) — matrix operand order repaired and checked by reading, not compiled, not run on hardware
//
#include <hip/hip_runtime.h>
#include <hip/hip_bf16.h>

// ---------------------------------------------------------------------------
// AMTreeGRU on gfx1250 (MI455X).
// Pixel-major ([P][C]) activation layout so WMMA A (pixels x K) and B
// (K x couts from [Cout][Cin] weights) fragments are contiguous 16B loads.
// Heavy math via v_wmma_f32_16x16x32_f16 with fp32 accumulation.
// Weight chunks staged into LDS double-buffered via the CDNA5 async
// global->LDS DMA path (ASYNCcnt + s_wait_asynccnt; sync fallback kept).
// B fragments double-buffered in registers at kk-group granularity so each
// WMMA group only waits dscnt<=8 (next group's loads), never a full drain.
// ---------------------------------------------------------------------------

typedef _Float16 h16_t;
typedef __attribute__((ext_vector_type(16))) _Float16     v16h;
typedef __attribute__((ext_vector_type(8)))  float        v8f;
typedef __attribute__((ext_vector_type(4)))  unsigned int v4u;
typedef int vi4_t __attribute__((vector_size(16)));   // type the async builtin wants

#define PIX    2704      // 52*52
#define DIMW   52
#define HID    256
#define NNODE  31
#define MTILES 169       // 2704/16
#define BROW   40        // padded LDS row stride (halves) per cout (32 + 8)
#define LDSH   (4 * 64 * BROW)   // halves per staging buffer (one 128-K chunk)

#if defined(__AMDGCN__) &&                                                     \
    __has_builtin(__builtin_amdgcn_global_load_async_to_lds_b128) &&           \
    __has_builtin(__builtin_amdgcn_s_wait_asynccnt)
#define USE_ASYNC 1
#else
#define USE_ASYNC 0
#endif

union F16Frag { v4u q[2]; v16h v; };

__device__ __forceinline__ v16h ld_frag(const h16_t* lo, const h16_t* hi) {
  F16Frag f;
  f.q[0] = *reinterpret_cast<const v4u*>(lo);
  f.q[1] = *reinterpret_cast<const v4u*>(hi);
  return f.v;
}

__device__ __forceinline__ v8f wmma16(v16h a, v16h b, v8f c) {
  return __builtin_amdgcn_wmma_f32_16x16x32_f16(false, a, false, b, (short)0, c,
                                                false, false);
}

__device__ __forceinline__ float apply_act(float x, int act) {
  if (act == 1) return (x > 20.f) ? x : log1pf(expf(x));   // softplus
  if (act == 2) return 1.f / (1.f + expf(-x));             // sigmoid
  if (act == 3) return tanhf(x);                           // tanh
  return x;
}

// copy one 8-half (16B) unit of a weight chunk into LDS
__device__ __forceinline__ void stage_unit(const h16_t* src, h16_t* dst) {
#if USE_ASYNC
  __builtin_amdgcn_global_load_async_to_lds_b128(
      (vi4_t*)const_cast<h16_t*>(src), (vi4_t*)dst, 0, 0);
#else
  *reinterpret_cast<v4u*>(dst) = *reinterpret_cast<const v4u*>(src);
#endif
}

__device__ __forceinline__ void stage_drain_and_sync() {
#if USE_ASYNC
  __builtin_amdgcn_s_wait_asynccnt(0);
#endif
  __syncthreads();
}

// ---------------------------------------------------------------------------
// Generic pixel-major GEMM:  out[P][N] = X[P][K] * Wt[N][K]^T (+bias, act)
// block = 128 threads = 4 waves; wave -> one 16-pixel M tile, 64 couts shared
// by the block.  Weight chunks (64 couts x 128 K) staged in LDS, dbl-buffered.
// grid = (ceil(MTILES/4), N/64).   K must be a multiple of 128.
// ---------------------------------------------------------------------------
__global__ __launch_bounds__(128) void gemm_pm(
    const h16_t* __restrict__ X, const h16_t* __restrict__ Wt, int K,
    const float* __restrict__ bias, float* __restrict__ outF,
    h16_t* __restrict__ outH, int ldo, int act)
{
  __shared__ __align__(16) h16_t Bs[2][LDSH];

  const int tid  = threadIdx.x;
  const int wave = tid >> 5;
  const int lane = tid & 31;
  const int l15  = lane & 15;
  const int mt   = blockIdx.x * 4 + wave;
  const bool mvalid = (mt < MTILES);
  const int p0   = mt << 4;
  int mrow = p0 + l15;
  if (mrow >= PIX) mrow = PIX - 1;          // clamp; stores guarded by mvalid
  const int n0   = blockIdx.y << 6;
  const int hi   = lane >> 4;

  const h16_t* arow = X + (size_t)mrow * K + hi * 8;

  // cooperative copy of one 64x128 weight chunk into LDS buffer `buf`
  auto stage = [&](int ks, int buf) {
#pragma unroll
    for (int i = 0; i < 8; ++i) {
      const int u    = tid + i * 128;        // 0..1023 (8-half units)
      const int cout = u >> 4;               // 0..63
      const int k8   = (u & 15) << 3;        // 0..120
      const int kk   = k8 >> 5, kl = k8 & 31;
      stage_unit(Wt + (size_t)(n0 + cout) * K + ks + k8,
                 &Bs[buf][(kk * 64 + cout) * BROW + kl]);
    }
  };

  v8f acc[4] = { {}, {}, {}, {} };
  const int nchunks = K >> 7;

  stage(0, 0);
  stage_drain_and_sync();

  for (int g = 0; g < nchunks; ++g) {
    // A fragments for this chunk (issued BEFORE next-chunk staging so the
    // WMMAs depend only on these loads)
    const h16_t* ab = arow + (g << 7);
    v16h a[4];
#pragma unroll
    for (int kk = 0; kk < 4; ++kk)
      a[kk] = ld_frag(ab + kk * 32, ab + kk * 32 + 16);

    if (g + 1 < nchunks) {
      __builtin_prefetch(arow + ((g + 1) << 7), 0, 1);
      stage((g + 1) << 7, (g + 1) & 1);
    }

    // 16 WMMAs; B fragments double-buffered per kk-group (4 frags ahead)
    const h16_t* bbase = &Bs[g & 1][0];
    auto ldB = [&](int kk, v16h* dst) {
#pragma unroll
      for (int j = 0; j < 4; ++j) {
        const h16_t* q = bbase + (kk * 64 + (j << 4) + l15) * BROW + hi * 8;
        dst[j] = ld_frag(q, q + 16);
      }
    };
    v16h b0[4], b1[4];
    ldB(0, b0);
#pragma unroll
    for (int kk = 0; kk < 4; ++kk) {
      v16h* cur = (kk & 1) ? b1 : b0;
      v16h* nxt = (kk & 1) ? b0 : b1;
      if (kk < 3) ldB(kk + 1, nxt);
#pragma unroll
      for (int j = 0; j < 4; ++j)
        acc[j] = wmma16(a[kk], cur[j], acc[j]);
    }
    stage_drain_and_sync();
  }

  if (mvalid) {
#pragma unroll
    for (int j = 0; j < 4; ++j) {
      const int cout = n0 + (j << 4) + l15;
      const float bv = bias ? bias[cout] : 0.f;
#pragma unroll
      for (int v = 0; v < 8; ++v) {
        const int p = p0 + v + hi * 8;
        const float x = apply_act(acc[j][v] + bv, act);
        const size_t o = (size_t)p * ldo + cout;
        if (outF) outF[o] = x;
        if (outH) outH[o] = (h16_t)x;
      }
    }
  }
}

// ---------------------------------------------------------------------------
// 3x3 conv, 512 in -> 256 out, SAME pad, implicit GEMM (K = 9 taps * 512).
// Xin: [P][512] f16 (Wy | S).  W9: [256][9*512] f16 ([cout][tap][cin]).
// out: [P][256] f32 with activation.  36 chunks of 128 K, LDS dbl-buffered.
// ---------------------------------------------------------------------------
__global__ __launch_bounds__(128) void conv3x3_pm(
    const h16_t* __restrict__ Xin, const h16_t* __restrict__ W9,
    const float* __restrict__ bias, float* __restrict__ outF, int act)
{
  __shared__ __align__(16) h16_t Bs[2][LDSH];

  const int tid  = threadIdx.x;
  const int wave = tid >> 5;
  const int lane = tid & 31;
  const int l15  = lane & 15;
  const int mt   = blockIdx.x * 4 + wave;
  const bool mvalid = (mt < MTILES);
  const int p0   = mt << 4;
  int mrow = p0 + l15;
  if (mrow >= PIX) mrow = PIX - 1;
  const int n0   = blockIdx.y << 6;
  const int hi   = lane >> 4;
  const int y    = mrow / DIMW;
  const int x    = mrow - y * DIMW;

  auto stage = [&](int ks, int buf) {
#pragma unroll
    for (int i = 0; i < 8; ++i) {
      const int u    = tid + i * 128;
      const int cout = u >> 4;
      const int k8   = (u & 15) << 3;
      const int kk   = k8 >> 5, kl = k8 & 31;
      stage_unit(W9 + (size_t)(n0 + cout) * 4608 + ks + k8,
                 &Bs[buf][(kk * 64 + cout) * BROW + kl]);
    }
  };

  v8f acc[4] = { {}, {}, {}, {} };

  stage(0, 0);
  stage_drain_and_sync();

  for (int g = 0; g < 36; ++g) {           // 9 taps * 4 chunks
    const int t  = g >> 2;
    const int dy = t / 3 - 1, dx = t % 3 - 1;
    const int yy = y + dy, xx = x + dx;
    const bool valid = (yy >= 0) & (yy < DIMW) & (xx >= 0) & (xx < DIMW);
    const h16_t* arow =
        Xin + (size_t)(yy * DIMW + xx) * 512 + ((g & 3) << 7) + hi * 8;

    v16h a[4] = {};
    if (valid) {
#pragma unroll
      for (int kk = 0; kk < 4; ++kk)
        a[kk] = ld_frag(arow + kk * 32, arow + kk * 32 + 16);
    }

    if (g + 1 < 36) {
      const int tn  = (g + 1) >> 2;
      const int yn  = y + tn / 3 - 1, xn = x + tn % 3 - 1;
      if ((yn >= 0) & (yn < DIMW) & (xn >= 0) & (xn < DIMW))
        __builtin_prefetch(Xin + (size_t)(yn * DIMW + xn) * 512 +
                               (((g + 1) & 3) << 7), 0, 1);
      stage(((g + 1) >> 2) * 512 + (((g + 1) & 3) << 7), (g + 1) & 1);
    }

    const h16_t* bbase = &Bs[g & 1][0];
    auto ldB = [&](int kk, v16h* dst) {
#pragma unroll
      for (int j = 0; j < 4; ++j) {
        const h16_t* q = bbase + (kk * 64 + (j << 4) + l15) * BROW + hi * 8;
        dst[j] = ld_frag(q, q + 16);
      }
    };
    v16h b0[4], b1[4];
    ldB(0, b0);
#pragma unroll
    for (int kk = 0; kk < 4; ++kk) {
      v16h* cur = (kk & 1) ? b1 : b0;
      v16h* nxt = (kk & 1) ? b0 : b1;
      if (kk < 3) ldB(kk + 1, nxt);
#pragma unroll
      for (int j = 0; j < 4; ++j)
        acc[j] = wmma16(a[kk], cur[j], acc[j]);
    }
    stage_drain_and_sync();
  }

  if (mvalid) {
#pragma unroll
    for (int j = 0; j < 4; ++j) {
      const int cout = n0 + (j << 4) + l15;
      const float bv = bias[cout];
#pragma unroll
      for (int v = 0; v < 8; ++v) {
        const int p = p0 + v + hi * 8;
        outF[(size_t)p * HID + cout] = apply_act(acc[j][v] + bv, act);
      }
    }
  }
}

// ------------------------------ prep kernels --------------------------------

__global__ void k_cvt(h16_t* dst, const float* src, int n) {
  int e = blockIdx.x * 256 + threadIdx.x;
  if (e < n) dst[e] = (h16_t)src[e];
}

// mconv_w [256][768] -> mw1 (k 0..255) f16, mw2 (k 256..511) f16
__global__ void k_cvt_mconv(h16_t* mw1, h16_t* mw2, const float* src) {
  int e = blockIdx.x * 256 + threadIdx.x;
  if (e >= 256 * 256) return;
  int c = e >> 8, k = e & 255;
  mw1[e] = (h16_t)src[(size_t)c * 768 + k];
  mw2[e] = (h16_t)src[(size_t)c * 768 + 256 + k];
}

// OIHW [256][512][3][3] -> [o][tap][cin] f16
__global__ void k_cvt_gru(h16_t* dst, const float* src) {
  int e = blockIdx.x * 256 + threadIdx.x;
  if (e >= 256 * 4608) return;
  int o = e / 4608, r = e % 4608, t = r >> 9, i = r & 511;
  dst[e] = (h16_t)src[((size_t)o * 512 + i) * 9 + t];
}

// vis [256][2704] f32 -> [2704][256] f16
__global__ void k_vispm(h16_t* dst, const float* vis) {
  int e = blockIdx.x * 256 + threadIdx.x;
  if (e >= PIX * HID) return;
  int p = e >> 8, c = e & 255;
  dst[e] = (h16_t)vis[(size_t)c * PIX + p];
}

// langc[idx][c] = mconv_w[c,512:768].lang[idx] + mconv_b[c]
// tlc[idx][j]   = theta_l_w[j].lang[idx] + theta_l_b[j] + theta_v_b[j]
__global__ void k_lang(const float* mconv_w, const float* mconv_b,
                       const float* tl_w, const float* tl_b, const float* tv_b,
                       const float* lang, float* langc, float* tlc) {
  int t = blockIdx.x * 256 + threadIdx.x;
  if (t >= NNODE * 384) return;
  int idx = t / 384, j = t % 384;
  const float* lv = lang + (size_t)idx * 256;
  if (j < 256) {
    float s = mconv_b[j];
    const float* wr = mconv_w + (size_t)j * 768 + 512;
    for (int k = 0; k < 256; ++k) s += wr[k] * lv[k];
    langc[(size_t)idx * 256 + j] = s;
  } else {
    int jj = j - 256;
    float s = tl_b[jj] + tv_b[jj];
    const float* wr = tl_w + (size_t)jj * 256;
    for (int k = 0; k < 256; ++k) s += wr[k] * lv[k];
    tlc[(size_t)idx * 128 + jj] = s;
  }
}

// --------------------------- per-node elementwise ---------------------------

// nf16 = A + att_in*B + langc[c];  att_in = 1 (leaf) or 1 - mean(attL,attR)
__global__ void k_nf(const float* A, const float* B, const float* attL,
                     const float* attR, const float* langc, h16_t* nf16) {
  int e = blockIdx.x * 256 + threadIdx.x;
  if (e >= PIX * HID) return;
  int p = e >> 8, c = e & 255;
  float ai = 1.f;
  if (attL) ai = 1.f - 0.5f * (attL[p] + attR[p]);
  nf16[e] = (h16_t)(A[e] + ai * B[e] + langc[c]);
}

// att[p] = sigmoid(psi_w . f[p] + psi_b)
__global__ void k_psi(const float* f, const float* psiw, const float* psib,
                      float* att) {
  int p = blockIdx.x * 256 + threadIdx.x;
  if (p >= PIX) return;
  float s = psib[0];
  const float* fr = f + (size_t)p * 128;
  for (int j = 0; j < 128; ++j) s += psiw[j] * fr[j];
  att[p] = 1.f / (1.f + expf(-s));
}

__global__ void k_g(const h16_t* nf16, const float* att, h16_t* g16) {
  int e = blockIdx.x * 256 + threadIdx.x;
  if (e >= PIX * HID) return;
  g16[e] = (h16_t)(att[e >> 8] * (float)nf16[e]);
}

__global__ void k_setS_sum(const h16_t* hl, const h16_t* hr, h16_t* convin) {
  int e = blockIdx.x * 256 + threadIdx.x;
  if (e >= PIX * HID) return;
  int p = e >> 8, c = e & 255;
  convin[(size_t)p * 512 + 256 + c] = (h16_t)((float)hl[e] + (float)hr[e]);
}

__global__ void k_setS_copy(const h16_t* src, h16_t* convin) {
  int e = blockIdx.x * 256 + threadIdx.x;
  if (e >= PIX * HID) return;
  int p = e >> 8, c = e & 255;
  convin[(size_t)p * 512 + 256 + c] = src[e];
}

__global__ void k_setS_rh(const float* rl, const float* rr, const h16_t* hl,
                          const h16_t* hr, h16_t* convin) {
  int e = blockIdx.x * 256 + threadIdx.x;
  if (e >= PIX * HID) return;
  int p = e >> 8, c = e & 255;
  convin[(size_t)p * 512 + 256 + c] =
      (h16_t)(rl[e] * (float)hl[e] + rr[e] * (float)hr[e]);
}

__global__ void k_zeroS(h16_t* convin) {
  int e = blockIdx.x * 256 + threadIdx.x;
  if (e >= PIX * HID) return;
  int p = e >> 8, c = e & 255;
  convin[(size_t)p * 512 + 256 + c] = (h16_t)0.f;
}

// h = (1-z)*ri + z*ch_sum ; optionally emit root output channel-major f32
__global__ void k_h(const float* z, const float* ri, const h16_t* hl,
                    const h16_t* hr, h16_t* hout, float* rootOut) {
  int e = blockIdx.x * 256 + threadIdx.x;
  if (e >= PIX * HID) return;
  float cs = 0.f;
  if (hl) cs = (float)hl[e] + (float)hr[e];
  float zz = z[e];
  float h = (1.f - zz) * ri[e] + zz * cs;
  hout[e] = (h16_t)h;
  if (rootOut) {
    int p = e >> 8, c = e & 255;
    rootOut[(size_t)c * PIX + p] = h;
  }
}

__global__ void k_attout(const float* att0, float* out) {
  int p = blockIdx.x * 256 + threadIdx.x;
  if (p < PIX) out[p] = att0[p];
}

// ---------------------------------------------------------------------------

extern "C" void kernel_launch(void* const* d_in, const int* in_sizes, int n_in,
                              void* d_out, int out_size, void* d_ws,
                              size_t ws_size, hipStream_t stream) {
  const float* vis      = (const float*)d_in[0];
  const float* lang     = (const float*)d_in[1];
  const float* mconv_w  = (const float*)d_in[2];
  const float* mconv_b  = (const float*)d_in[3];
  const float* reset_w  = (const float*)d_in[4];
  const float* reset_b  = (const float*)d_in[5];
  const float* update_w = (const float*)d_in[6];
  const float* update_b = (const float*)d_in[7];
  const float* output_w = (const float*)d_in[8];
  const float* output_b = (const float*)d_in[9];
  const float* thv_w    = (const float*)d_in[10];
  const float* thv_b    = (const float*)d_in[11];
  const float* thl_w    = (const float*)d_in[12];
  const float* thl_b    = (const float*)d_in[13];
  const float* psi_w    = (const float*)d_in[14];
  const float* psi_b    = (const float*)d_in[15];
  const float* W_w      = (const float*)d_in[16];
  const float* W_b      = (const float*)d_in[17];
  // d_in[18] adj: tree is the fixed full binary tree (child = 2i+1, 2i+2)

  // workspace bump allocator
  char* wp = (char*)d_ws;
  auto alloc = [&](size_t n) -> void* {
    void* p = (void*)wp;
    wp += (n + 255) & ~(size_t)255;
    return p;
  };

  h16_t* h16   = (h16_t*)alloc((size_t)NNODE * PIX * HID * 2);  // node hidden f16
  float* attB  = (float*)alloc((size_t)NNODE * PIX * 4);        // node attention
  h16_t* vispm = (h16_t*)alloc((size_t)PIX * HID * 2);
  h16_t* mw1   = (h16_t*)alloc((size_t)256 * 256 * 2);
  h16_t* mw2   = (h16_t*)alloc((size_t)256 * 256 * 2);
  h16_t* thv16 = (h16_t*)alloc((size_t)128 * 256 * 2);
  h16_t* Ww16  = (h16_t*)alloc((size_t)256 * 256 * 2);
  h16_t* w9r   = (h16_t*)alloc((size_t)256 * 4608 * 2);
  h16_t* w9u   = (h16_t*)alloc((size_t)256 * 4608 * 2);
  h16_t* w9o   = (h16_t*)alloc((size_t)256 * 4608 * 2);
  float* langc = (float*)alloc((size_t)NNODE * 256 * 4);
  float* tlc   = (float*)alloc((size_t)NNODE * 128 * 4);
  float* Apm   = (float*)alloc((size_t)PIX * HID * 4);
  float* Bvpm  = (float*)alloc((size_t)PIX * HID * 4);
  h16_t* nf16  = (h16_t*)alloc((size_t)PIX * HID * 2);
  float* fbuf  = (float*)alloc((size_t)PIX * 128 * 4);
  h16_t* g16   = (h16_t*)alloc((size_t)PIX * HID * 2);
  h16_t* convin = (h16_t*)alloc((size_t)PIX * 512 * 2);
  float* zbuf  = (float*)alloc((size_t)PIX * HID * 4);
  float* ribuf = (float*)alloc((size_t)PIX * HID * 4);
  float* rlbuf = (float*)alloc((size_t)PIX * HID * 4);
  float* rrbuf = (float*)alloc((size_t)PIX * HID * 4);

  const int pcB = (PIX * HID) / 256;  // 2704 blocks of 256
  const dim3 gemmGrid256(43, 4), gemmGrid128(43, 2), convGrid(43, 4);

  // ---- prep: f16 weights, rearrangements, language constants ----
  k_cvt_mconv<<<(256 * 256 + 255) / 256, 256, 0, stream>>>(mw1, mw2, mconv_w);
  k_cvt<<<(128 * 256 + 255) / 256, 256, 0, stream>>>(thv16, thv_w, 128 * 256);
  k_cvt<<<(256 * 256 + 255) / 256, 256, 0, stream>>>(Ww16, W_w, 256 * 256);
  k_cvt_gru<<<(256 * 4608 + 255) / 256, 256, 0, stream>>>(w9r, reset_w);
  k_cvt_gru<<<(256 * 4608 + 255) / 256, 256, 0, stream>>>(w9u, update_w);
  k_cvt_gru<<<(256 * 4608 + 255) / 256, 256, 0, stream>>>(w9o, output_w);
  k_vispm<<<pcB, 256, 0, stream>>>(vispm, vis);
  k_lang<<<(NNODE * 384 + 255) / 256, 256, 0, stream>>>(
      mconv_w, mconv_b, thl_w, thl_b, thv_b, lang, langc, tlc);

  // node-independent mconv halves: A = W1.vis, B = W2.vis   (f32 out)
  gemm_pm<<<gemmGrid256, 128, 0, stream>>>(vispm, mw1, 256, nullptr, Apm,
                                           nullptr, HID, 0);
  gemm_pm<<<gemmGrid256, 128, 0, stream>>>(vispm, mw2, 256, nullptr, Bvpm,
                                           nullptr, HID, 0);

  // ---- bottom-up tree traversal (children before parents) ----
  for (int idx = NNODE - 1; idx >= 0; --idx) {
    const bool leaf = (2 * idx + 1 >= NNODE);
    const int l = 2 * idx + 1, r = 2 * idx + 2;
    const h16_t* hl = leaf ? nullptr : h16 + (size_t)l * PIX * HID;
    const h16_t* hr = leaf ? nullptr : h16 + (size_t)r * PIX * HID;

    // nf = A + att_in*B + langc[idx]
    k_nf<<<pcB, 256, 0, stream>>>(Apm, Bvpm,
                                  leaf ? nullptr : attB + (size_t)l * PIX,
                                  leaf ? nullptr : attB + (size_t)r * PIX,
                                  langc + (size_t)idx * 256, nf16);
    // f = softplus(theta_v(nf) + theta_l(lang) + biases)   [P][128] f32
    gemm_pm<<<gemmGrid128, 128, 0, stream>>>(nf16, thv16, 256,
                                             tlc + (size_t)idx * 128, fbuf,
                                             nullptr, 128, 1 /*softplus*/);
    // att = sigmoid(psi(f))
    k_psi<<<(PIX + 255) / 256, 256, 0, stream>>>(fbuf, psi_w, psi_b,
                                                 attB + (size_t)idx * PIX);
    // g = att * nf
    k_g<<<pcB, 256, 0, stream>>>(nf16, attB + (size_t)idx * PIX, g16);
    // Wy = W.g + W_b  -> convin[:,0:256] f16 (stride 512)
    gemm_pm<<<gemmGrid256, 128, 0, stream>>>(g16, Ww16, 256, W_b, nullptr,
                                             convin, 512, 0);

    if (leaf) {
      k_zeroS<<<pcB, 256, 0, stream>>>(convin);
      conv3x3_pm<<<convGrid, 128, 0, stream>>>(convin, w9u, update_b, zbuf, 2);
      conv3x3_pm<<<convGrid, 128, 0, stream>>>(convin, w9o, output_b, ribuf, 3);
      k_h<<<pcB, 256, 0, stream>>>(zbuf, ribuf, nullptr, nullptr,
                                   h16 + (size_t)idx * PIX * HID,
                                   idx == 0 ? (float*)d_out : nullptr);
    } else {
      // z = sigmoid(conv(Wy | hl+hr))
      k_setS_sum<<<pcB, 256, 0, stream>>>(hl, hr, convin);
      conv3x3_pm<<<convGrid, 128, 0, stream>>>(convin, w9u, update_b, zbuf, 2);
      // r_l, r_r = sigmoid(conv(Wy | h_child))
      k_setS_copy<<<pcB, 256, 0, stream>>>(hl, convin);
      conv3x3_pm<<<convGrid, 128, 0, stream>>>(convin, w9r, reset_b, rlbuf, 2);
      k_setS_copy<<<pcB, 256, 0, stream>>>(hr, convin);
      conv3x3_pm<<<convGrid, 128, 0, stream>>>(convin, w9r, reset_b, rrbuf, 2);
      // ri = tanh(conv(Wy | r_l*h_l + r_r*h_r))
      k_setS_rh<<<pcB, 256, 0, stream>>>(rlbuf, rrbuf, hl, hr, convin);
      conv3x3_pm<<<convGrid, 128, 0, stream>>>(convin, w9o, output_b, ribuf, 3);
      // h = (1-z)*ri + z*(hl+hr)
      k_h<<<pcB, 256, 0, stream>>>(zbuf, ribuf, hl, hr,
                                   h16 + (size_t)idx * PIX * HID,
                                   idx == 0 ? (float*)d_out : nullptr);
    }
  }

  // root attention -> tail of d_out
  k_attout<<<(PIX + 255) / 256, 256, 0, stream>>>(attB,
                                                  (float*)d_out + (size_t)HID * PIX);
}
